// Attention_module_52166672777937
// MI455X (gfx1250) — matrix operand order repaired and checked by reading, not compiled, not run on hardware
//
#include <hip/hip_runtime.h>
#include <hip/hip_bf16.h>

#define BATCH 32
#define SEQ   1000
#define DM    512
#define NH    8
#define DH    64
#define ROWS  (BATCH * SEQ)   // 32000

typedef __attribute__((ext_vector_type(16))) __bf16 v16bf;
typedef __attribute__((ext_vector_type(8)))  float  v8f;

// ---------------------------------------------------------------- helpers
__device__ __forceinline__ float pos_enc(int t, int e) {
    // pe[t, 2i] = sin(t * exp(-(2i) ln(1e4)/512)); pe[t, 2i+1] = cos(...)
    const float c = -9.210340371976184f / 512.0f;   // -ln(10000)/d_model
    float div = __expf((float)(e & ~1) * c);
    float arg = (float)t * div;
    return (e & 1) ? __cosf(arg) : __sinf(arg);
}

// A fragment (16x32 bf16, MxK): lane half selects K groups {kSel..+7, kSel+16..+23}
__device__ __forceinline__ v16bf fragA(const __bf16* p) {
    union { v16bf v; uint4 q[2]; } u;
    u.q[0] = *(const uint4*)(p);
    u.q[1] = *(const uint4*)(p + 16);
    return u.v;
}
// B fragment (32x16 bf16, KxN): lane = column; lane half selects K groups {kSel..+7, kSel+16..+23}
__device__ __forceinline__ v16bf fragB(const __bf16* p) {
    union { v16bf v; uint4 q[2]; } u;
    u.q[0] = *(const uint4*)(p);
    u.q[1] = *(const uint4*)(p + 16);
    return u.v;
}

// ---------------------------------------------------------------- prep: x = emb[data] + pe  (bf16)
__global__ void prep_x(const int* __restrict__ data, const float* __restrict__ emb,
                       __bf16* __restrict__ xb) {
    int r = blockIdx.x;                 // 0..31999
    int t = r % SEQ;
    int ch = data[r];
    const float* er = emb + (size_t)ch * DM;
    __bf16* xr = xb + (size_t)r * DM;
    for (int e = threadIdx.x; e < DM; e += blockDim.x)
        xr[e] = (__bf16)(er[e] + pos_enc(t, e));
}

__global__ void cvt_bf16(const float* __restrict__ src, __bf16* __restrict__ dst, int n) {
    int i = blockIdx.x * blockDim.x + threadIdx.x;
    if (i < n) dst[i] = (__bf16)src[i];
}

// ---------------------------------------------------------------- fused K/V projection GEMM (WMMA)
// out[r,n] = sum_e x[r,e] * W[n,e] + b[n]   for W in {Wk, Wv}, shared A fragments.
// Block: 8 waves; wave w -> rows [bx*128 + 16w, +16), cols [by*64, +64). Grid (250, 8).
__global__ __launch_bounds__(256) void kv_gemm(
        const __bf16* __restrict__ xb, const __bf16* __restrict__ wkb,
        const __bf16* __restrict__ wvb, const float* __restrict__ bk,
        const float* __restrict__ bv, __bf16* __restrict__ kb, __bf16* __restrict__ vb) {
    const int lane = threadIdx.x & 31;
    const int wave = threadIdx.x >> 5;
    const int mtile = blockIdx.x * 8 + wave;          // 16-row tile index
    const int col0  = blockIdx.y * 64;
    const int kSelA = (lane >> 4) << 3;               // 0 or 8
    const int kSelB = (lane >> 4) << 3;               // 0 or 8

    const __bf16* Arow = xb + (size_t)(mtile * 16 + (lane & 15)) * DM;
    const __bf16* Bk[4];
    const __bf16* Bv[4];
#pragma unroll
    for (int nt = 0; nt < 4; ++nt) {
        int n = col0 + nt * 16 + (lane & 15);
        Bk[nt] = wkb + (size_t)n * DM;
        Bv[nt] = wvb + (size_t)n * DM;
    }

    const v8f vzero = {0.f, 0.f, 0.f, 0.f, 0.f, 0.f, 0.f, 0.f};
    v8f ak[4], av[4];
#pragma unroll
    for (int nt = 0; nt < 4; ++nt) { ak[nt] = vzero; av[nt] = vzero; }

    for (int k = 0; k < DM; k += 32) {
        v16bf a = fragA(Arow + k + kSelA);
#pragma unroll
        for (int nt = 0; nt < 4; ++nt) {
            v16bf b0 = fragB(Bk[nt] + k + kSelB);
            ak[nt] = __builtin_amdgcn_wmma_f32_16x16x32_bf16(
                         false, a, false, b0, (short)0, ak[nt], false, false);
            v16bf b1 = fragB(Bv[nt] + k + kSelB);
            av[nt] = __builtin_amdgcn_wmma_f32_16x16x32_bf16(
                         false, a, false, b1, (short)0, av[nt], false, false);
        }
    }

    // D layout: lane = col (lane&15); VGPR i -> row (lane>=16 ? 8 : 0) + i
    const int srow0 = mtile * 16 + ((lane >> 4) << 3);
#pragma unroll
    for (int nt = 0; nt < 4; ++nt) {
        int col = col0 + nt * 16 + (lane & 15);
        float bbk = bk[col], bbv = bv[col];
#pragma unroll
        for (int i = 0; i < 8; ++i) {
            size_t off = (size_t)(srow0 + i) * DM + col;
            kb[off] = (__bf16)(ak[nt][i] + bbk);
            vb[off] = (__bf16)(av[nt][i] + bbv);
        }
    }
}

// ---------------------------------------------------------------- Q at last valid row per batch
__global__ __launch_bounds__(256) void q_proj(
        const int* __restrict__ data, const int* __restrict__ lengths,
        const float* __restrict__ emb, const float* __restrict__ Wq,
        const float* __restrict__ bq, float* __restrict__ qf) {
    __shared__ float sx[DM];
    int b = blockIdx.x;
    int p = lengths[b]; if (p < 1) p = 1; if (p > SEQ) p = SEQ;
    int t = p - 1;
    int ch = data[b * SEQ + t];
    for (int e = threadIdx.x; e < DM; e += blockDim.x)
        sx[e] = emb[(size_t)ch * DM + e] + pos_enc(t, e);
    __syncthreads();
    for (int n = threadIdx.x; n < DM; n += blockDim.x) {
        float s = bq[n];
        const float* wr = Wq + (size_t)n * DM;
        for (int e = 0; e < DM; ++e) s += sx[e] * wr[e];
        qf[b * DM + n] = s;
    }
}

// ---------------------------------------------------------------- single-query attention + MLP head
// One block per batch; wave = head. Scores -> softmax -> ctx -> W1 -> W2 -> mean -> leaky.
__global__ __launch_bounds__(256) void attn_head(
        const int* __restrict__ lengths, const __bf16* __restrict__ kb,
        const __bf16* __restrict__ vb, const float* __restrict__ qf,
        const float* __restrict__ W1, const float* __restrict__ b1,
        const float* __restrict__ W2, const float* __restrict__ b2,
        float* __restrict__ out) {
    __shared__ float sc[NH][1024];
    __shared__ float sq[DM];
    __shared__ float sctx[DM];
    __shared__ float sh1[DM];
    __shared__ float so[NH];

    const int b = blockIdx.x;
    const int tid = threadIdx.x, lane = tid & 31, h = tid >> 5;
    int p = lengths[b]; if (p < 1) p = 1; if (p > SEQ) p = SEQ;

    sq[tid]       = qf[b * DM + tid];
    sq[tid + 256] = qf[b * DM + tid + 256];
    __syncthreads();

    const float scale = 0.125f;                        // 1/sqrt(64)
    const __bf16* kbb = kb + (size_t)b * SEQ * DM + h * DH;
    const __bf16* vbb = vb + (size_t)b * SEQ * DM + h * DH;

    // phase 1: scores for this head's keys (lane-strided over t)
    float lmax = -3.0e38f;
    for (int t = lane; t < p; t += 32) {
        const __bf16* kr = kbb + (size_t)t * DM;
        float s = 0.f;
        for (int d = 0; d < DH; ++d) s += sq[h * DH + d] * (float)kr[d];
        s *= scale;
        sc[h][t] = s;
        lmax = fmaxf(lmax, s);
    }
    for (int off = 16; off; off >>= 1) lmax = fmaxf(lmax, __shfl_xor(lmax, off, 32));
    float lsum = 0.f;
    for (int t = lane; t < p; t += 32) {
        float e = __expf(sc[h][t] - lmax);
        sc[h][t] = e;
        lsum += e;
    }
    for (int off = 16; off; off >>= 1) lsum += __shfl_xor(lsum, off, 32);
    const float inv = 1.0f / lsum;
    __syncthreads();                                   // cross-lane sc visibility

    // phase 2: ctx = (sum_t w_t * v_t) / sum  ; lane handles dims d, d+32
    const int d0 = lane, d1 = lane + 32;
    float a0 = 0.f, a1 = 0.f;
    for (int t = 0; t < p; ++t) {
        float w = sc[h][t];
        const __bf16* vr = vbb + (size_t)t * DM;
        a0 += w * (float)vr[d0];
        a1 += w * (float)vr[d1];
    }
    sctx[h * DH + d0] = a0 * inv;
    sctx[h * DH + d1] = a1 * inv;
    __syncthreads();

    // MLP layer 1 (leaky_relu, slope 0.01)
    for (int n = tid; n < DM; n += 256) {
        float s = b1[n];
        const float* wr = W1 + (size_t)n * DM;
        for (int e = 0; e < DM; ++e) s += sctx[e] * wr[e];
        sh1[n] = s > 0.f ? s : 0.01f * s;
    }
    __syncthreads();

    // MLP layer 2 (relu) -> mean -> leaky_relu
    if (tid < NH) {
        float s = b2[tid];
        const float* wr = W2 + (size_t)tid * DM;
        for (int e = 0; e < DM; ++e) s += sh1[e] * wr[e];
        so[tid] = fmaxf(s, 0.f);
    }
    __syncthreads();
    if (tid == 0) {
        float m = 0.f;
        for (int j = 0; j < NH; ++j) m += so[j];
        m *= (1.0f / NH);
        out[b] = m > 0.f ? m : 0.01f * m;
    }
}

// ---------------------------------------------------------------- launch
extern "C" void kernel_launch(void* const* d_in, const int* in_sizes, int n_in,
                              void* d_out, int out_size, void* d_ws, size_t ws_size,
                              hipStream_t stream) {
    const int*   data    = (const int*)  d_in[0];
    const int*   lengths = (const int*)  d_in[1];
    const float* emb     = (const float*)d_in[2];
    const float* Wq      = (const float*)d_in[3];
    const float* bq      = (const float*)d_in[4];
    const float* Wk      = (const float*)d_in[5];
    const float* bk      = (const float*)d_in[6];
    const float* Wv      = (const float*)d_in[7];
    const float* bv      = (const float*)d_in[8];
    const float* W1      = (const float*)d_in[9];
    const float* b1      = (const float*)d_in[10];
    const float* W2      = (const float*)d_in[11];
    const float* b2      = (const float*)d_in[12];

    char* ws = (char*)d_ws;
    size_t off = 0;
    auto alloc = [&](size_t bytes) -> void* {
        void* p = ws + off;
        off += (bytes + 255) & ~(size_t)255;
        return p;
    };
    __bf16* xb  = (__bf16*)alloc((size_t)ROWS * DM * 2);   // 32.8 MB
    __bf16* wkb = (__bf16*)alloc((size_t)DM * DM * 2);
    __bf16* wvb = (__bf16*)alloc((size_t)DM * DM * 2);
    __bf16* kb  = (__bf16*)alloc((size_t)ROWS * DM * 2);   // 32.8 MB
    __bf16* vb  = (__bf16*)alloc((size_t)ROWS * DM * 2);   // 32.8 MB
    float*  qf  = (float*) alloc((size_t)BATCH * DM * 4);

    prep_x<<<ROWS, 256, 0, stream>>>(data, emb, xb);
    cvt_bf16<<<(DM * DM + 255) / 256, 256, 0, stream>>>(Wk, wkb, DM * DM);
    cvt_bf16<<<(DM * DM + 255) / 256, 256, 0, stream>>>(Wv, wvb, DM * DM);
    kv_gemm<<<dim3(ROWS / 128, DM / 64), 256, 0, stream>>>(xb, wkb, wvb, bk, bv, kb, vb);
    q_proj<<<BATCH, 256, 0, stream>>>(data, lengths, emb, Wq, bq, qf);
    attn_head<<<BATCH, 256, 0, stream>>>(lengths, kb, vb, qf, W1, b1, W2, b2, (float*)d_out);
}
